// TEBD_39384850104987
// MI455X (gfx1250) — matrix operand order repaired and checked by reading, not compiled, not run on hardware
//
#include <hip/hip_runtime.h>
#include <hip/hip_bf16.h>
#include <math.h>

// ---------------------------------------------------------------------------
// RealNVP coupling stack on MI455X (gfx1250, wave32, WMMA).
// Dominant cost: per layer GEMM  h1(262144x256) @ W2(256x256)  in bf16 WMMA
// with f32 accumulation (v_wmma_f32_16x16x32_bf16).  Everything is L2
// resident (~4MB total), so the kernel is matrix-core/transcendental bound.
// ---------------------------------------------------------------------------

typedef __attribute__((ext_vector_type(16))) __bf16 v16bf;
typedef __attribute__((ext_vector_type(8)))  float  v8f;

#define BATCH   2048
#define NBIT    256
#define HID     256
#define NLAYER  6
#define ROWS    (BATCH * (NBIT / 2))   // 262144 coupling rows
#define TM      128                    // rows per workgroup
#define KCH     8                      // K chunks of 32 (K = 256)
#define NB      16                     // N blocks of 16 (N = 256)
#define FRAG_PER_LAYER (KCH * NB * 32 * 16)   // bf16 elems per layer = 65536*... (8*16*32*16)

__device__ __forceinline__ float fast_tanh(float v) {
#if __has_builtin(__builtin_amdgcn_tanhf)
    return __builtin_amdgcn_tanhf(v);        // gfx1250 V_TANH_F32
#elif __has_builtin(__builtin_amdgcn_tanh_f32)
    return __builtin_amdgcn_tanh_f32(v);
#else
    return tanhf(v);
#endif
}

// ---------------------------------------------------------------------------
// One-time repack of W2 (L,H,H f32 row-major) into bf16 B-fragment-major
// layout: [layer][kk][nb][lane][16].  Matches the ISA 32x16 bf16 B layout:
//   lanes 0-15 : N = lane,    K = kkBase + (v>=4?16:0) + (v&3)*2 + h   (v = elem>>1, h = elem&1)
//   lanes 16-31: N = lane-16, K = kkBase + (v>=4?16:0) + 8 + (v&3)*2 + h
// so in the GEMM each lane loads its whole fragment as one contiguous v16bf.
// ---------------------------------------------------------------------------
__global__ void pack_w2(const float* __restrict__ W2, __bf16* __restrict__ w2f) {
    int t = blockIdx.x * blockDim.x + threadIdx.x;
    if (t >= NLAYER * FRAG_PER_LAYER) return;
    int e    = t & 15;         int r = t >> 4;
    int lane = r & 31;         r >>= 5;
    int nb   = r & 15;         r >>= 4;
    int kk   = r & 7;          int layer = r >> 3;
    int v = e >> 1, h = e & 1;
    int k = kk * 32 + ((v >= 4) ? 16 : 0) + ((lane & 16) ? 8 : 0) + (v & 3) * 2 + h;
    int n = nb * 16 + (lane & 15);
    w2f[t] = (__bf16)W2[((size_t)layer * HID + k) * HID + n];
}

// ---------------------------------------------------------------------------
// One coupling layer, in place on x.  Grid: ROWS/TM blocks of 256 threads
// (8 wave32s).  Wave w owns rows [blk*128 + 16w, +16) x all 256 columns.
// ---------------------------------------------------------------------------
__global__ __launch_bounds__(256)
void coupling_layer(float* __restrict__ x,
                    const __bf16* __restrict__ w2f,   // this layer's fragments
                    const float* __restrict__ W1,     // [H]
                    const float* __restrict__ b1,     // [H]
                    const float* __restrict__ b2,     // [H]
                    const float* __restrict__ W3,     // [H][2] row-major
                    const float* __restrict__ b3,     // [2]
                    int odd)
{
    const int lane = threadIdx.x & 31;
    const int wave = threadIdx.x >> 5;
    const int l15  = lane & 15;
    const bool hiHalf = lane >= 16;

    // x0 for this wave's A-fragment row (both lane halves hold the same M row)
    const int m_base = blockIdx.x * TM;
    const int mA = m_base + wave * 16 + l15;
    const int bA = mA >> 7;                // batch index (128 pairs per batch row)
    const int iA = mA & 127;               // pair index
    const int idx0 = odd ? (2 * iA + 1) : (2 * iA);
    const float x0 = x[bA * NBIT + idx0];

    v8f acc[NB];
#pragma unroll
    for (int nb = 0; nb < NB; ++nb) acc[nb] = (v8f){0.f,0.f,0.f,0.f,0.f,0.f,0.f,0.f};

    const v16bf* __restrict__ bfrags = (const v16bf*)w2f;   // [(kk*NB+nb)*32+lane]

    for (int kk = 0; kk < KCH; ++kk) {
        // warm next K-chunk's B fragments into cache (global_prefetch_b8)
        __builtin_prefetch(&bfrags[(((kk + 1) & (KCH - 1)) * NB) * 32 + lane], 0, 0);

        // Synthesize A fragment: h1 = tanh(x0*W1 + b1), ISA 16x32 bf16 A layout:
        //  VGPR v, lo/hi h:  K = kk*32 + (v>=4?16:0) + (hiHalf?8:0) + (v&3)*2 + h
        v16bf a;
#pragma unroll
        for (int e = 0; e < 16; ++e) {
            const int v = e >> 1, hb = e & 1;
            const int k = kk * 32 + ((v >= 4) ? 16 : 0) + (hiHalf ? 8 : 0) + (v & 3) * 2 + hb;
            a[e] = (__bf16)fast_tanh(fmaf(x0, W1[k], b1[k]));
        }

#pragma unroll
        for (int nb = 0; nb < NB; ++nb) {
            const v16bf bf = bfrags[(kk * NB + nb) * 32 + lane];
            acc[nb] = __builtin_amdgcn_wmma_f32_16x16x32_bf16(
                false, a, false, bf, (short)0, acc[nb], false, false);
        }
    }

    // Epilogue in registers.  C layout: acc[nb] VGPR r, lane l ->
    //   row = r + (hiHalf ? 8 : 0), col = nb*16 + l15.
    // h2 = tanh(acc + b2); partial s,t = h2 @ W3 per lane.
    float s_part[8], t_part[8];
#pragma unroll
    for (int r = 0; r < 8; ++r) { s_part[r] = 0.f; t_part[r] = 0.f; }

#pragma unroll
    for (int nb = 0; nb < NB; ++nb) {
        const int n   = nb * 16 + l15;
        const float bias = b2[n];
        const float w3s = W3[2 * n], w3t = W3[2 * n + 1];
#pragma unroll
        for (int r = 0; r < 8; ++r) {
            const float h2 = fast_tanh(acc[nb][r] + bias);
            s_part[r] = fmaf(h2, w3s, s_part[r]);
            t_part[r] = fmaf(h2, w3t, t_part[r]);
        }
    }

    // Reduce over the 16 lanes of each half-wave (xor masks stay inside a half)
#pragma unroll
    for (int r = 0; r < 8; ++r) {
#pragma unroll
        for (int m = 8; m >= 1; m >>= 1) {
            s_part[r] += __shfl_xor(s_part[r], m);
            t_part[r] += __shfl_xor(t_part[r], m);
        }
    }

    // Route row (lane&7)'s sums to this lane via unrolled cndmask select.
    const int rsel = lane & 7;
    float s = s_part[0], t = t_part[0];
#pragma unroll
    for (int r = 1; r < 8; ++r) {
        s = (rsel == r) ? s_part[r] : s;
        t = (rsel == r) ? t_part[r] : t;
    }
    s += b3[0];
    t += b3[1];

    // Writers: lanes 0-7 -> rows 0-7 of the wave tile, lanes 16-23 -> rows 8-15.
    if (l15 < 8) {
        const int m_local = wave * 16 + (hiHalf ? 8 : 0) + rsel;
        const int mg = m_base + m_local;
        const int b  = mg >> 7, i = mg & 127;
        const int idx1 = odd ? ((2 * i + 2) & (NBIT - 1)) : (2 * i + 1);
        const float yold = x[b * NBIT + idx1];
        x[b * NBIT + idx1] = fmaf(yold, expf(s), t);
    }
}

extern "C" void kernel_launch(void* const* d_in, const int* in_sizes, int n_in,
                              void* d_out, int out_size, void* d_ws, size_t ws_size,
                              hipStream_t stream) {
    (void)in_sizes; (void)n_in; (void)out_size; (void)ws_size;
    const float* x_in = (const float*)d_in[0];
    const float* W1   = (const float*)d_in[1];   // (L,1,H)
    const float* b1   = (const float*)d_in[2];   // (L,H)
    const float* W2   = (const float*)d_in[3];   // (L,H,H)
    const float* b2   = (const float*)d_in[4];   // (L,H)
    const float* W3   = (const float*)d_in[5];   // (L,H,2)
    const float* b3   = (const float*)d_in[6];   // (L,2)
    float*  x   = (float*)d_out;
    __bf16* w2f = (__bf16*)d_ws;                 // 6 * 128KB = 768KB

    // Work in place on d_out; inputs stay untouched.
    hipMemcpyAsync(x, x_in, (size_t)BATCH * NBIT * sizeof(float),
                   hipMemcpyDeviceToDevice, stream);

    const int packTotal = NLAYER * FRAG_PER_LAYER;
    pack_w2<<<(packTotal + 255) / 256, 256, 0, stream>>>(W2, w2f);

    for (int l = 0; l < NLAYER; ++l) {
        coupling_layer<<<ROWS / TM, 256, 0, stream>>>(
            x,
            w2f + (size_t)l * FRAG_PER_LAYER,
            W1 + l * HID, b1 + l * HID, b2 + l * HID,
            W3 + l * HID * 2, b3 + l * 2,
            l & 1);
    }
}
